// FlowEstimator_70411693851142
// MI455X (gfx1250) — matrix operand order repaired and checked by reading, not compiled, not run on hardware
//
#include <hip/hip_runtime.h>
#include <math.h>

// ---------------------------------------------------------------------------
// CDNA5 (gfx1250) FlowEstimator.
// All convs + the global correlation run as implicit GEMMs on
// V_WMMA_F32_16X16X32_F16 with operands pre-packed into WMMA-native f16
// layouts.  Each wave computes a 32x32 output block (2x2 register blocking):
// per K-chunk, 8 unconditional global_load_b128 feed 4 WMMAs.
// ---------------------------------------------------------------------------

typedef __attribute__((ext_vector_type(16))) _Float16 v16h;
typedef __attribute__((ext_vector_type(8)))  _Float16 h8;
typedef __attribute__((ext_vector_type(8)))  float    v8f;

// ---------------------------------------------------------------------------
// Pack conv weights OIHW f32 -> f16 [Mpad][KK][Cpad], zero-filled.
// ---------------------------------------------------------------------------
__global__ void k_pack_w(const float* __restrict__ w, int Cout, int Ctot,
                         int KK, int Cpad, int Mpad, _Float16* __restrict__ out)
{
    size_t total = (size_t)Mpad * KK * Cpad;
    size_t idx = (size_t)blockIdx.x * blockDim.x + threadIdx.x;
    if (idx >= total) return;
    int c = (int)(idx % Cpad);
    int t = (int)(idx / Cpad);
    int p = t % KK;
    int m = t / KK;
    float v = (m < Cout && c < Ctot) ? w[((size_t)m * Ctot + c) * KK + p] : 0.f;
    out[idx] = (_Float16)v;
}

// ---------------------------------------------------------------------------
// Pack (concat of up to two) NCHW f32 activations -> padded NHWC f16
// [(H+2p)*(W+2p)][Cpad], zero halo + zero channel padding.
// ---------------------------------------------------------------------------
__global__ void k_pack_in(const float* __restrict__ in0, int C0,
                          const float* __restrict__ in1, int C1,
                          int H, int W, int pad, int Cpad,
                          _Float16* __restrict__ out)
{
    const int Wp = W + 2 * pad, Hp = H + 2 * pad;
    size_t total = (size_t)Hp * Wp * Cpad;
    size_t idx = (size_t)blockIdx.x * blockDim.x + threadIdx.x;
    if (idx >= total) return;
    int c = (int)(idx % Cpad);
    int q = (int)(idx / Cpad);
    int ix = q % Wp - pad;
    int iy = q / Wp - pad;
    float v = 0.f;
    if ((unsigned)iy < (unsigned)H && (unsigned)ix < (unsigned)W) {
        if (c < C0)           v = in0[(size_t)c * H * W + iy * W + ix];
        else if (c < C0 + C1) v = in1[(size_t)(c - C0) * H * W + iy * W + ix];
    }
    out[idx] = (_Float16)v;
}

// Transpose NCHW f32 -> [n][c] f16 (C multiple of 32).
__global__ void k_pack_t(const float* __restrict__ x, int C, int N,
                         _Float16* __restrict__ out)
{
    size_t idx = (size_t)blockIdx.x * blockDim.x + threadIdx.x;
    if (idx >= (size_t)N * C) return;
    int c = (int)(idx % C);
    int n = (int)(idx / C);
    out[idx] = (_Float16)x[(size_t)c * N + n];
}

// ---------------------------------------------------------------------------
// Packed implicit-GEMM conv, 2x2 register blocking.
// One wave32 computes a 32(ch) x 32(pix) block as four 16x16 WMMA tiles.
//   inpk : padded NHWC f16 [(H+2p)*(W+2p)][Cpad]
//   wpk  : f16 [Mpad][KK][Cpad]   (Mpad multiple of 32, zero-filled)
// Inner chunk: 8 x global_load_b128 -> 4 x v_wmma_f32_16x16x32_f16.
// H*W multiple of 32; gridDim = (H*W/32, Mpad/32).
// act: 0 none, 1 relu, 2 sigmoid.
// ---------------------------------------------------------------------------
__global__ void k_conv_packed(const _Float16* __restrict__ inpk,
                              int H, int W, int pad, int Cpad,
                              const _Float16* __restrict__ wpk,
                              const float* __restrict__ bias,
                              int Cout, int ks, int act,
                              float* __restrict__ out)
{
    const int lane  = threadIdx.x & 31;
    const int col   = lane & 15;
    const int hi    = lane >> 4;
    const int tileN = blockIdx.x * 32;
    const int tileM = blockIdx.y * 32;
    const int Wp    = W + 2 * pad;
    const int Npix  = H * W;
    const int KK    = ks * ks;

    const int n0 = tileN + col,      n1 = n0 + 16;
    const int oy0 = n0 / W, ox0 = n0 % W;
    const int oy1 = n1 / W, ox1 = n1 % W;
    const int mA0 = tileM + col;                 // < Mpad by construction
    const int mA1 = mA0 + 16;

    v8f acc00 = {}, acc01 = {}, acc10 = {}, acc11 = {};

    for (int p = 0; p < KK; ++p) {
        const int ky = p / ks, kx = p - ky * ks;
        const _Float16* brow0 = inpk + ((size_t)(oy0 + ky) * Wp + (ox0 + kx)) * Cpad;
        const _Float16* brow1 = inpk + ((size_t)(oy1 + ky) * Wp + (ox1 + kx)) * Cpad;
        const _Float16* arow0 = wpk + ((size_t)mA0 * KK + p) * Cpad;
        const _Float16* arow1 = wpk + ((size_t)mA1 * KK + p) * Cpad;
        for (int c0 = 0; c0 < Cpad; c0 += 32) {
            // A fragments: element e -> c = c0 + (e/8)*16 + hi*8 + (e%8)
            h8 a0l = *(const h8*)(arow0 + c0 + hi * 8);
            h8 a0h = *(const h8*)(arow0 + c0 + 16 + hi * 8);
            h8 a1l = *(const h8*)(arow1 + c0 + hi * 8);
            h8 a1h = *(const h8*)(arow1 + c0 + 16 + hi * 8);
            // B fragments: element e -> c = c0 + (e/8)*16 + hi*8 + (e%8)
            h8 b0l = *(const h8*)(brow0 + c0 + hi * 8);
            h8 b0h = *(const h8*)(brow0 + c0 + 16 + hi * 8);
            h8 b1l = *(const h8*)(brow1 + c0 + hi * 8);
            h8 b1h = *(const h8*)(brow1 + c0 + 16 + hi * 8);
            v16h A0 = __builtin_shufflevector(a0l, a0h, 0,1,2,3,4,5,6,7,8,9,10,11,12,13,14,15);
            v16h A1 = __builtin_shufflevector(a1l, a1h, 0,1,2,3,4,5,6,7,8,9,10,11,12,13,14,15);
            v16h B0 = __builtin_shufflevector(b0l, b0h, 0,1,2,3,4,5,6,7,8,9,10,11,12,13,14,15);
            v16h B1 = __builtin_shufflevector(b1l, b1h, 0,1,2,3,4,5,6,7,8,9,10,11,12,13,14,15);
            acc00 = __builtin_amdgcn_wmma_f32_16x16x32_f16(false, A0, false, B0, (short)0, acc00, false, false);
            acc01 = __builtin_amdgcn_wmma_f32_16x16x32_f16(false, A0, false, B1, (short)0, acc01, false, false);
            acc10 = __builtin_amdgcn_wmma_f32_16x16x32_f16(false, A1, false, B0, (short)0, acc10, false, false);
            acc11 = __builtin_amdgcn_wmma_f32_16x16x32_f16(false, A1, false, B1, (short)0, acc11, false, false);
        }
    }

#pragma unroll
    for (int r = 0; r < 8; ++r) {
        int m0 = tileM + r + (hi << 3);
        int m1 = m0 + 16;
        if (m0 < Cout) {
            float bsv = bias ? bias[m0] : 0.f;
            float v0 = acc00[r] + bsv, v1 = acc01[r] + bsv;
            if (act == 1)      { v0 = fmaxf(v0, 0.f); v1 = fmaxf(v1, 0.f); }
            else if (act == 2) { v0 = 1.f / (1.f + __expf(-v0)); v1 = 1.f / (1.f + __expf(-v1)); }
            out[(size_t)m0 * Npix + n0] = v0;
            out[(size_t)m0 * Npix + n1] = v1;
        }
        if (m1 < Cout) {
            float bsv = bias ? bias[m1] : 0.f;
            float v0 = acc10[r] + bsv, v1 = acc11[r] + bsv;
            if (act == 1)      { v0 = fmaxf(v0, 0.f); v1 = fmaxf(v1, 0.f); }
            else if (act == 2) { v0 = 1.f / (1.f + __expf(-v0)); v1 = 1.f / (1.f + __expf(-v1)); }
            out[(size_t)m1 * Npix + n0] = v0;
            out[(size_t)m1 * Npix + n1] = v1;
        }
    }
}

// ---------------------------------------------------------------------------
// GroupNorm (per group over Cg*HW) + exact GELU.  One block per group.
// ---------------------------------------------------------------------------
__global__ void k_gn_gelu(const float* __restrict__ x,
                          const float* __restrict__ g,
                          const float* __restrict__ bt,
                          int C, int groups, int HW,
                          float* __restrict__ out)
{
    const int grp  = blockIdx.x;
    const int Cg   = C / groups;
    const int base = grp * Cg * HW;
    const int cnt  = Cg * HW;
    __shared__ float s1[256], s2[256];

    float a = 0.f, b2 = 0.f;
    for (int i = threadIdx.x; i < cnt; i += blockDim.x) {
        float v = x[base + i];
        a += v; b2 += v * v;
    }
    s1[threadIdx.x] = a; s2[threadIdx.x] = b2;
    __syncthreads();
    for (int s = blockDim.x >> 1; s > 0; s >>= 1) {
        if (threadIdx.x < s) {
            s1[threadIdx.x] += s1[threadIdx.x + s];
            s2[threadIdx.x] += s2[threadIdx.x + s];
        }
        __syncthreads();
    }
    const float mu  = s1[0] / (float)cnt;
    const float var = s2[0] / (float)cnt - mu * mu;
    const float inv = rsqrtf(var + 1e-5f);

    for (int i = threadIdx.x; i < cnt; i += blockDim.x) {
        int c   = grp * Cg + i / HW;
        float v = (x[base + i] - mu) * inv * g[c] + bt[c];
        v = 0.5f * v * (1.f + erff(v * 0.70710678118654752440f));
        out[base + i] = v;
    }
}

// Per-pixel channel L2 normalization.
__global__ void k_l2norm(const float* __restrict__ x, int C, int N,
                         float* __restrict__ out)
{
    int n = blockIdx.x * blockDim.x + threadIdx.x;
    if (n >= N) return;
    float s = 0.f;
    for (int c = 0; c < C; ++c) { float v = x[(size_t)c * N + n]; s += v * v; }
    float inv = 1.f / fmaxf(sqrtf(s), 1e-6f);
    for (int c = 0; c < C; ++c) out[(size_t)c * N + n] = x[(size_t)c * N + n] * inv;
}

// Row softmax of clipped/scaled correlation + expected coords -> flow.
__global__ void k_softmax_expect(const float* __restrict__ corr, int H, int W,
                                 float scale, float* __restrict__ flow)
{
    const int N = H * W;
    const int n = blockIdx.x;
    const float* row = corr + (size_t)n * N;
    __shared__ float r0[256], r1[256], r2[256];

    float mx = -1e30f;
    for (int m = threadIdx.x; m < N; m += blockDim.x) {
        float v = fminf(fmaxf(row[m] * scale, -50.f), 50.f);
        mx = fmaxf(mx, v);
    }
    r0[threadIdx.x] = mx; __syncthreads();
    for (int s = blockDim.x >> 1; s > 0; s >>= 1) {
        if (threadIdx.x < s)
            r0[threadIdx.x] = fmaxf(r0[threadIdx.x], r0[threadIdx.x + s]);
        __syncthreads();
    }
    mx = r0[0]; __syncthreads();

    float se = 0.f, sx = 0.f, sy = 0.f;
    for (int m = threadIdx.x; m < N; m += blockDim.x) {
        float v = fminf(fmaxf(row[m] * scale, -50.f), 50.f);
        float e = __expf(v - mx);
        se += e;
        sx += e * (float)(m % W);
        sy += e * (float)(m / W);
    }
    r0[threadIdx.x] = se; r1[threadIdx.x] = sx; r2[threadIdx.x] = sy;
    __syncthreads();
    for (int s = blockDim.x >> 1; s > 0; s >>= 1) {
        if (threadIdx.x < s) {
            r0[threadIdx.x] += r0[threadIdx.x + s];
            r1[threadIdx.x] += r1[threadIdx.x + s];
            r2[threadIdx.x] += r2[threadIdx.x + s];
        }
        __syncthreads();
    }
    if (threadIdx.x == 0) {
        float inv = 1.f / r0[0];
        flow[n]     = r1[0] * inv - (float)(n % W);
        flow[N + n] = r2[0] * inv - (float)(n / W);
    }
}

// Convex-mask upsample (up=2).
__global__ void k_convex_combine(const float* __restrict__ mask,
                                 const float* __restrict__ flowc,
                                 int Hc, int Wc, float* __restrict__ flowf)
{
    const int Wf = Wc * 2;
    const int Nf = Hc * 2 * Wf;
    int idx = blockIdx.x * blockDim.x + threadIdx.x;
    if (idx >= Nf) return;
    int y = idx / Wf, x = idx % Wf;
    int yc = y >> 1, xc = x >> 1;
    int sub = ((y & 1) << 1) | (x & 1);
    const int Nc = Hc * Wc;
    const int pc = yc * Wc + xc;

    float w[9]; float mx = -1e30f;
#pragma unroll
    for (int t = 0; t < 9; ++t) {
        w[t] = mask[(size_t)(t * 4 + sub) * Nc + pc];
        mx = fmaxf(mx, w[t]);
    }
    float s = 0.f;
#pragma unroll
    for (int t = 0; t < 9; ++t) { w[t] = __expf(w[t] - mx); s += w[t]; }
    float inv = 1.f / s;

#pragma unroll
    for (int ch = 0; ch < 2; ++ch) {
        float acc = 0.f;
#pragma unroll
        for (int t = 0; t < 9; ++t) {
            int dy = t / 3 - 1, dx = t % 3 - 1;
            int iy = yc + dy, ix = xc + dx;
            float f = (iy >= 0 && iy < Hc && ix >= 0 && ix < Wc)
                        ? flowc[(size_t)ch * Nc + iy * Wc + ix] : 0.f;
            acc += w[t] * f;
        }
        flowf[(size_t)ch * Nf + idx] = 2.f * acc * inv;
    }
}

__global__ void k_add(float* __restrict__ dst, const float* __restrict__ src, int n)
{
    int i = blockIdx.x * blockDim.x + threadIdx.x;
    if (i < n) dst[i] += src[i];
}

__global__ void k_pack_merge(const float* __restrict__ flr,
                             const float* __restrict__ frl,
                             const float* __restrict__ fs1,
                             int N, float* __restrict__ out)
{
    int n = blockIdx.x * blockDim.x + threadIdx.x;
    if (n >= N) return;
    float a0 = flr[n], a1 = flr[N + n];
    float b0 = frl[n], b1 = frl[N + n];
    out[0 * (size_t)N + n] = a0;
    out[1 * (size_t)N + n] = a1;
    out[2 * (size_t)N + n] = b0;
    out[3 * (size_t)N + n] = b1;
    out[4 * (size_t)N + n] = 0.5f * (fabsf(a0) + fabsf(a1));
    out[5 * (size_t)N + n] = 0.5f * (fabsf(b0) + fabsf(b1));
    for (int c = 0; c < 32; ++c)
        out[(size_t)(6 + c) * N + n] = fs1[(size_t)c * N + n];
}

__global__ void k_finalize(const float* __restrict__ flr,
                           const float* __restrict__ frl,
                           const float* __restrict__ delta,
                           const float* __restrict__ clr,
                           const float* __restrict__ crl,
                           int N, float* __restrict__ out)
{
    int n = blockIdx.x * blockDim.x + threadIdx.x;
    if (n >= N) return;
    out[0 * (size_t)N + n] = 0.5f * flr[n]     + delta[0 * (size_t)N + n];
    out[1 * (size_t)N + n] = 0.5f * flr[N + n] + delta[1 * (size_t)N + n];
    out[2 * (size_t)N + n] = 0.5f * frl[n]     + delta[2 * (size_t)N + n];
    out[3 * (size_t)N + n] = 0.5f * frl[N + n] + delta[3 * (size_t)N + n];
    out[4 * (size_t)N + n] = clr[n];
    out[5 * (size_t)N + n] = crl[n];
}

// ---------------------------------------------------------------------------
// Host orchestration
// ---------------------------------------------------------------------------
enum {
    IN_FS1 = 0, IN_FS2, IN_FS3, IN_FS4, IN_FS5,
    IN_FT1, IN_FT2, IN_FT3, IN_FT4, IN_FT5,
    I_PROJ_W, I_GN_G, I_GN_B,
    I_R8U_W1, I_R8U_B1, I_R8U_W2, I_R8U_B2,
    I_R8R_C1, I_R8R_C1B, I_R8R_C2, I_R8R_C2B,
    I_R8R_H1, I_R8R_H1B, I_R8R_H2, I_R8R_H2B,
    I_R4U_W1, I_R4U_B1, I_R4U_W2, I_R4U_B2,
    I_R4R_C1, I_R4R_C1B, I_R4R_C2, I_R4R_C2B,
    I_R4R_H1, I_R4R_H1B, I_R4R_H2, I_R4R_H2B,
    I_R2U_W1, I_R2U_B1, I_R2U_W2, I_R2U_B2,
    I_R2R_C1, I_R2R_C1B, I_R2R_C2, I_R2R_C2B,
    I_R2R_H1, I_R2R_H1B, I_R2R_H2, I_R2R_H2B,
    I_FUP_W1, I_FUP_B1, I_FUP_W2, I_FUP_B2,
    I_CF_W1, I_CF_B1, I_CF_W2, I_CF_B2,
    I_MF_W1, I_MF_B1, I_MF_W2, I_MF_B2, I_MF_W3, I_MF_B3
};

// Distinct conv weight tensors (packed once per launch).
enum {
    CW_PROJ = 0,
    CW_R8U1, CW_R8U2, CW_R8R_C1, CW_R8R_C2, CW_R8R_H1, CW_R8R_H2,
    CW_R4U1, CW_R4U2, CW_R4R_C1, CW_R4R_C2, CW_R4R_H1, CW_R4R_H2,
    CW_R2U1, CW_R2U2, CW_R2R_C1, CW_R2R_C2, CW_R2R_H1, CW_R2R_H2,
    CW_FUP1, CW_FUP2, CW_CF1, CW_CF2, CW_MF1, CW_MF2, CW_MF3,
    CW_COUNT
};
struct ConvWDesc { int widx; int bidx; int Cout; int Ctot; int ks; };
static const ConvWDesc CWL[CW_COUNT] = {
    { I_PROJ_W, -1,        128, 160, 1 },
    { I_R8U_W1, I_R8U_B1,  256, 160, 3 }, { I_R8U_W2, I_R8U_B2, 36, 256, 1 },
    { I_R8R_C1, I_R8R_C1B, 128, 130, 3 }, { I_R8R_C2, I_R8R_C2B, 128, 128, 3 },
    { I_R8R_H1, I_R8R_H1B,  64, 128, 3 }, { I_R8R_H2, I_R8R_H2B,   2,  64, 3 },
    { I_R4U_W1, I_R4U_B1,  256, 128, 3 }, { I_R4U_W2, I_R4U_B2, 36, 256, 1 },
    { I_R4R_C1, I_R4R_C1B,  96,  98, 3 }, { I_R4R_C2, I_R4R_C2B,  96,  96, 3 },
    { I_R4R_H1, I_R4R_H1B,  64,  96, 3 }, { I_R4R_H2, I_R4R_H2B,   2,  64, 3 },
    { I_R2U_W1, I_R2U_B1,  256,  96, 3 }, { I_R2U_W2, I_R2U_B2, 36, 256, 1 },
    { I_R2R_C1, I_R2R_C1B,  64,  66, 3 }, { I_R2R_C2, I_R2R_C2B,  64,  64, 3 },
    { I_R2R_H1, I_R2R_H1B,  64,  64, 3 }, { I_R2R_H2, I_R2R_H2B,   2,  64, 3 },
    { I_FUP_W1, I_FUP_B1,  256,  64, 3 }, { I_FUP_W2, I_FUP_B2, 36, 256, 1 },
    { I_CF_W1,  I_CF_B1,    64,  34, 3 }, { I_CF_W2,  I_CF_B2,    1,  64, 1 },
    { I_MF_W1,  I_MF_B1,   128,  38, 3 }, { I_MF_W2,  I_MF_B2,  128, 128, 3 },
    { I_MF_W3,  I_MF_B3,     4, 128, 3 },
};
static inline int ceil32(int x) { return (x + 31) & ~31; }

// f32 workspace layout (float offsets)
static const size_t O_MASKH  = 0;
static const size_t O_MASK36 = 7500000;
static const size_t O_T1     = 8600000;
static const size_t O_T2     = 10600000;
static const size_t O_D      = 12600000;
static const size_t O_F16    = 12700000;
static const size_t O_F8     = 12710000;
static const size_t O_F4     = 12720000;
static const size_t O_F2     = 12740000;
static const size_t O_CORR   = 12800000;
static const size_t O_INP    = 0;
static const size_t O_H2     = 0;
static const size_t REGION2  = 15000000;
static const size_t PERS     = 29700000;
static const size_t P_RAW = PERS + 0;
static const size_t P_PA  = PERS + 57344;
static const size_t P_PB  = PERS + 114688;
static const size_t P_NA  = PERS + 172032;
static const size_t P_NB  = PERS + 229376;
static const size_t P_FLR = PERS + 286720;
static const size_t P_FRL = PERS + 516096;
static const size_t P_CLR = PERS + 745472;
static const size_t P_CRL = PERS + 860160;
// f16 regions (expressed as float offsets; half pointers derived)
static const size_t WPK_F   = 30700000;   // packed weights (<= 4M halves)
static const size_t QPK_F   = 32700000;   // 57,344 halves
static const size_t KPK_F   = 32750000;   // 57,344 halves
static const size_t STAGE_F = 32800000;   // <= 15.2M halves
// total ~40.4M floats (~162 MB)

struct Ctx {
    const float* const* P;
    float* ws;
    _Float16* wpk[CW_COUNT];
    int cpad[CW_COUNT];
    int mpad[CW_COUNT];
    _Float16* stage;
    hipStream_t st;
};

static void conv_run(const Ctx& cx, int cw,
                     const float* in0, int C0, const float* in1, int C1,
                     int H, int W, int act, float* out)
{
    const ConvWDesc& d = CWL[cw];
    const int ks = d.ks, pad = ks >> 1;
    const int Cpad = cx.cpad[cw];
    const int Hp = H + 2 * pad, Wp = W + 2 * pad;
    size_t tot = (size_t)Hp * Wp * Cpad;
    k_pack_in<<<(unsigned)((tot + 255) / 256), 256, 0, cx.st>>>(
        in0, C0, in1, C1, H, W, pad, Cpad, cx.stage);
    const float* bias = (d.bidx >= 0) ? cx.P[d.bidx] : nullptr;
    dim3 g((unsigned)((H * W) / 32), (unsigned)(cx.mpad[cw] / 32));
    k_conv_packed<<<g, 32, 0, cx.st>>>(cx.stage, H, W, pad, Cpad,
                                       cx.wpk[cw], bias, d.Cout, ks, act, out);
}

static void run_direction(const Ctx& cx,
                          const float* f1, const float* f2, const float* f3,
                          const float* f4, const float* f5,
                          const _Float16* qpk, const _Float16* kpk,
                          float* flow_full, float* conf)
{
    float* ws = cx.ws;
    hipStream_t st = cx.st;
    // --- 1/16 global matching: corr = qpk (448x128) x kpk^T via packed GEMM
    {
        dim3 g(448 / 32, 448 / 32);
        k_conv_packed<<<g, 32, 0, st>>>(kpk, 1, 448, 0, 128, qpk, nullptr,
                                        448, 1, 0, ws + O_CORR);
        k_softmax_expect<<<448, 256, 0, st>>>(ws + O_CORR, 16, 28,
                                              0.88388347648318440f, ws + O_F16);
    }
    // --- convex up 1/16 -> 1/8 (feat f5 @16x28)
    conv_run(cx, CW_R8U1, f5, 160, nullptr, 0, 16, 28, 1, ws + O_MASKH);
    conv_run(cx, CW_R8U2, ws + O_MASKH, 256, nullptr, 0, 16, 28, 0, ws + O_MASK36);
    k_convex_combine<<<(1792 + 255) / 256, 256, 0, st>>>(ws + O_MASK36, ws + O_F16, 16, 28, ws + O_F8);
    // --- refine @1/8 (feat f4 @32x56)
    conv_run(cx, CW_R8R_C1, f4, 128, ws + O_F8, 2, 32, 56, 1, ws + O_T1);
    conv_run(cx, CW_R8R_C2, ws + O_T1, 128, nullptr, 0, 32, 56, 0, ws + O_T2);
    conv_run(cx, CW_R8R_H1, ws + O_T2, 128, nullptr, 0, 32, 56, 1, ws + O_T1);
    conv_run(cx, CW_R8R_H2, ws + O_T1, 64, nullptr, 0, 32, 56, 0, ws + O_D);
    k_add<<<(2 * 1792 + 255) / 256, 256, 0, st>>>(ws + O_F8, ws + O_D, 2 * 1792);
    // --- convex up 1/8 -> 1/4
    conv_run(cx, CW_R4U1, f4, 128, nullptr, 0, 32, 56, 1, ws + O_MASKH);
    conv_run(cx, CW_R4U2, ws + O_MASKH, 256, nullptr, 0, 32, 56, 0, ws + O_MASK36);
    k_convex_combine<<<(7168 + 255) / 256, 256, 0, st>>>(ws + O_MASK36, ws + O_F8, 32, 56, ws + O_F4);
    // --- refine @1/4 (feat f3 @64x112)
    conv_run(cx, CW_R4R_C1, f3, 96, ws + O_F4, 2, 64, 112, 1, ws + O_T1);
    conv_run(cx, CW_R4R_C2, ws + O_T1, 96, nullptr, 0, 64, 112, 0, ws + O_T2);
    conv_run(cx, CW_R4R_H1, ws + O_T2, 96, nullptr, 0, 64, 112, 1, ws + O_T1);
    conv_run(cx, CW_R4R_H2, ws + O_T1, 64, nullptr, 0, 64, 112, 0, ws + O_D);
    k_add<<<(2 * 7168 + 255) / 256, 256, 0, st>>>(ws + O_F4, ws + O_D, 2 * 7168);
    // --- convex up 1/4 -> 1/2
    conv_run(cx, CW_R2U1, f3, 96, nullptr, 0, 64, 112, 1, ws + O_MASKH);
    conv_run(cx, CW_R2U2, ws + O_MASKH, 256, nullptr, 0, 64, 112, 0, ws + O_MASK36);
    k_convex_combine<<<(28672 + 255) / 256, 256, 0, st>>>(ws + O_MASK36, ws + O_F4, 64, 112, ws + O_F2);
    // --- refine @1/2 (feat f2 @128x224)
    conv_run(cx, CW_R2R_C1, f2, 64, ws + O_F2, 2, 128, 224, 1, ws + O_T1);
    conv_run(cx, CW_R2R_C2, ws + O_T1, 64, nullptr, 0, 128, 224, 0, ws + O_T2);
    conv_run(cx, CW_R2R_H1, ws + O_T2, 64, nullptr, 0, 128, 224, 1, ws + O_T1);
    conv_run(cx, CW_R2R_H2, ws + O_T1, 64, nullptr, 0, 128, 224, 0, ws + O_D);
    k_add<<<(2 * 28672 + 255) / 256, 256, 0, st>>>(ws + O_F2, ws + O_D, 2 * 28672);
    // --- convex up 1/2 -> full
    conv_run(cx, CW_FUP1, f2, 64, nullptr, 0, 128, 224, 1, ws + O_MASKH);
    conv_run(cx, CW_FUP2, ws + O_MASKH, 256, nullptr, 0, 128, 224, 0, ws + O_MASK36);
    k_convex_combine<<<(114688 + 255) / 256, 256, 0, st>>>(ws + O_MASK36, ws + O_F2, 128, 224, flow_full);
    // --- confidence head @full (feat f1 @256x448)
    conv_run(cx, CW_CF1, f1, 32, flow_full, 2, 256, 448, 1, ws + O_MASKH);
    conv_run(cx, CW_CF2, ws + O_MASKH, 64, nullptr, 0, 256, 448, 2, conf);
}

extern "C" void kernel_launch(void* const* d_in, const int* in_sizes, int n_in,
                              void* d_out, int out_size, void* d_ws, size_t ws_size,
                              hipStream_t stream)
{
    (void)in_sizes; (void)n_in; (void)out_size; (void)ws_size;
    const float* const* P = (const float* const*)d_in;
    float* ws  = (float*)d_ws;
    float* out = (float*)d_out;
    hipStream_t st = stream;

    Ctx cx;
    cx.P = P; cx.ws = ws; cx.st = st;
    cx.stage = (_Float16*)(ws + STAGE_F);

    // ---- pack all conv weights into WMMA layout (Mpad rounded to 32) -------
    {
        _Float16* wbase = (_Float16*)(ws + WPK_F);
        size_t cur = 0;
        for (int i = 0; i < CW_COUNT; ++i) {
            const ConvWDesc& d = CWL[i];
            const int KK = d.ks * d.ks;
            const int Cpad = ceil32(d.Ctot);
            const int Mpad = ceil32(d.Cout);
            cx.cpad[i] = Cpad; cx.mpad[i] = Mpad;
            cx.wpk[i]  = wbase + cur;
            size_t sz = (size_t)Mpad * KK * Cpad;
            k_pack_w<<<(unsigned)((sz + 255) / 256), 256, 0, st>>>(
                P[d.widx], d.Cout, d.Ctot, KK, Cpad, Mpad, cx.wpk[i]);
            cur += sz;
        }
    }

    // ---- shared projection @1/16: conv1x1(160->128) + GN(8) + GELU ---------
    conv_run(cx, CW_PROJ, P[IN_FS5], 160, nullptr, 0, 16, 28, 0, ws + P_RAW);
    k_gn_gelu<<<8, 256, 0, st>>>(ws + P_RAW, P[I_GN_G], P[I_GN_B], 128, 8, 448, ws + P_PA);
    conv_run(cx, CW_PROJ, P[IN_FT5], 160, nullptr, 0, 16, 28, 0, ws + P_RAW);
    k_gn_gelu<<<8, 256, 0, st>>>(ws + P_RAW, P[I_GN_G], P[I_GN_B], 128, 8, 448, ws + P_PB);
    k_l2norm<<<2, 256, 0, st>>>(ws + P_PA, 128, 448, ws + P_NA);
    k_l2norm<<<2, 256, 0, st>>>(ws + P_PB, 128, 448, ws + P_NB);
    // transpose normalized features to [pixel][128] f16 for the WMMA GEMM
    _Float16* qpk = (_Float16*)(ws + QPK_F);
    _Float16* kpk = (_Float16*)(ws + KPK_F);
    k_pack_t<<<(448 * 128 + 255) / 256, 256, 0, st>>>(ws + P_NA, 128, 448, qpk);
    k_pack_t<<<(448 * 128 + 255) / 256, 256, 0, st>>>(ws + P_NB, 128, 448, kpk);

    // ---- both directions ---------------------------------------------------
    run_direction(cx, P[IN_FS1], P[IN_FS2], P[IN_FS3], P[IN_FS4], P[IN_FS5],
                  qpk, kpk, ws + P_FLR, ws + P_CLR);
    run_direction(cx, P[IN_FT1], P[IN_FT2], P[IN_FT3], P[IN_FT4], P[IN_FT5],
                  kpk, qpk, ws + P_FRL, ws + P_CRL);

    // ---- merge head @full res ---------------------------------------------
    const int N = 256 * 448;
    k_pack_merge<<<(N + 255) / 256, 256, 0, st>>>(ws + P_FLR, ws + P_FRL,
                                                  P[IN_FS1], N, ws + O_INP);
    conv_run(cx, CW_MF1, ws + O_INP, 38, nullptr, 0, 256, 448, 1, ws + REGION2);
    conv_run(cx, CW_MF2, ws + REGION2, 128, nullptr, 0, 256, 448, 1, ws + O_H2);
    conv_run(cx, CW_MF3, ws + O_H2, 128, nullptr, 0, 256, 448, 0, ws + REGION2);
    k_finalize<<<(N + 255) / 256, 256, 0, st>>>(ws + P_FLR, ws + P_FRL, ws + REGION2,
                                                ws + P_CLR, ws + P_CRL, N, out);
}
